// Block_14010183320003
// MI455X (gfx1250) — matrix operand order repaired and checked by reading, not compiled, not run on hardware
//
#include <hip/hip_runtime.h>
#include <hip/hip_bf16.h>
#include <math.h>

#define BATCH 2
#define CDIM  62
#define NHEAD 2
#define DHEAD 31
#define HDIM  256
#define WDIM  256
#define NPIX  (HDIM*WDIM)          // 65536
#define RTOT  (BATCH*NPIX)         // 131072
#define MDIM  (4*CDIM)             // 248
#define CP    64                   // padded C
#define MP    256                  // padded M
#define C2P   128                  // padded 2C
#define KCONV (9*C2P)              // 1152 (padded conv K)

typedef __attribute__((ext_vector_type(16))) _Float16 v16h;
typedef __attribute__((ext_vector_type(8)))  _Float16 v8h;
typedef __attribute__((ext_vector_type(8)))  float    v8f;

__device__ __forceinline__ float gelu_exact(float x){
    return 0.5f * x * (1.0f + erff(x * 0.7071067811865475f));
}

__device__ __forceinline__ v8f wmma16(v16h a, v16h b, v8f c){
    return __builtin_amdgcn_wmma_f32_16x16x32_f16(false, a, false, b, (short)0, c,
                                                  false, false);
}

__device__ __forceinline__ v16h combine(v8h lo, v8h hi){
    v16h r;
#pragma unroll
    for (int i = 0; i < 8; ++i) { r[i] = lo[i]; r[8 + i] = hi[i]; }
    return r;
}

// ---------------------------------------------------------------------------
// GEMM: out[R x Co] = A[R x KP](f16, padded) * Wt[CoP x KP](f16, transposed,
// zero-padded). Fragments = unconditional aligned b128 loads.
// Block = 256 thr (8 waves); wave = one 16-row x 16-col tile.
// ---------------------------------------------------------------------------
__global__ void gemm_h(const _Float16* __restrict__ A, const _Float16* __restrict__ Wt,
                       const float* __restrict__ bias, const float* __restrict__ resid,
                       float* __restrict__ outF, _Float16* __restrict__ outH,
                       int KP, int Co, int strideH, int headpad, int dogelu)
{
    const int lane = threadIdx.x & 31;
    const int wave = threadIdx.x >> 5;
    const int row0 = (blockIdx.x * 8 + wave) * 16;
    const int col  = blockIdx.y * 16 + (lane & 15);
    const int arow = row0 + (lane & 15);
    const int akb  = (lane >> 4) * 16;
    const int bkb  = (lane >> 4) * 16;

    const _Float16* ap = A  + (size_t)arow * KP;
    const _Float16* bp = Wt + (size_t)col  * KP;

    v8f acc = {0.f,0.f,0.f,0.f,0.f,0.f,0.f,0.f};
    for (int k0 = 0; k0 < KP; k0 += 32) {
        v16h a = combine(*(const v8h*)(ap + k0 + akb),
                         *(const v8h*)(ap + k0 + akb + 8));
        v16h b = combine(*(const v8h*)(bp + k0 + bkb),
                         *(const v8h*)(bp + k0 + bkb + 8));
        acc = wmma16(a, b, acc);
    }

    const int mg = (lane >> 4) * 8;
    const int cc = blockIdx.y * 16 + (lane & 15);
    if (cc >= Co) return;
    const float bv = bias ? bias[cc] : 0.f;
    const int hc = headpad ? (cc + (cc >= DHEAD ? 1 : 0)) : cc;
#pragma unroll
    for (int r = 0; r < 8; ++r) {
        const int rr = row0 + mg + r;
        float v = acc[r] + bv;
        if (dogelu) v = gelu_exact(v);
        if (resid) v += resid[(size_t)rr * Co + cc];
        if (outF) outF[(size_t)rr * Co + cc] = v;
        if (outH) outH[(size_t)rr * strideH + hc] = (_Float16)v;
    }
}

// ---------------------------------------------------------------------------
// attn[b,h,d,e] = sum_n kn[d,n]*qn[e,n]; split-K over pixels through LDS.
// qnh/knh are head-padded f16 [pix][64] (slot h*32+d). Block=128thr=4 waves,
// wave w -> (d-tile=w>>1, e-tile=w&1); stage = 128 pixels transposed in LDS.
// ---------------------------------------------------------------------------
#define APIX 128
__global__ void attn_partial(const _Float16* __restrict__ knh,
                             const _Float16* __restrict__ qnh,
                             float* __restrict__ acc, int rows_per_block)
{
    __shared__ __align__(16) _Float16 lK[32][APIX + 8];
    __shared__ __align__(16) _Float16 lQ[32][APIX + 8];
    const int lane = threadIdx.x & 31;
    const int wave = threadIdx.x >> 5;
    const int dt = wave >> 1, et = wave & 1;
    const int bh = blockIdx.y;
    const int b = bh / NHEAD, h = bh % NHEAD;
    const int r0 = blockIdx.x * rows_per_block;
    const int akb = (lane >> 4) * 16;
    const int bkb = (lane >> 4) * 16;
    const int dm = dt * 16 + (lane & 15);
    const int en = et * 16 + (lane & 15);
    const int t = threadIdx.x;             // stage pixel 0..127

    v8f acc8 = {0.f,0.f,0.f,0.f,0.f,0.f,0.f,0.f};
    for (int s = 0; s < rows_per_block; s += APIX) {
        const size_t g = ((size_t)b * NPIX + r0 + s + t) * CP + h * 32;
#pragma unroll
        for (int c8 = 0; c8 < 4; ++c8) {
            const v8h kv = *(const v8h*)(knh + g + c8 * 8);
            const v8h qv = *(const v8h*)(qnh + g + c8 * 8);
#pragma unroll
            for (int j = 0; j < 8; ++j) {
                lK[c8 * 8 + j][t] = kv[j];
                lQ[c8 * 8 + j][t] = qv[j];
            }
        }
        __syncthreads();
#pragma unroll
        for (int kk = 0; kk < APIX; kk += 32) {
            v16h a = combine(*(const v8h*)&lK[dm][kk + akb],
                             *(const v8h*)&lK[dm][kk + akb + 8]);
            v16h bb = combine(*(const v8h*)&lQ[en][kk + bkb],
                              *(const v8h*)&lQ[en][kk + bkb + 8]);
            acc8 = wmma16(a, bb, acc8);
        }
        __syncthreads();
    }
    const int mg = (lane >> 4) * 8;
#pragma unroll
    for (int r = 0; r < 8; ++r) {
        const int dd = dt * 16 + mg + r;
        const int ee = et * 16 + (lane & 15);
        atomicAdd(&acc[((size_t)bh * 32 + dd) * 32 + ee], acc8[r]);
    }
}

// softmax over e; writes zero-padded row-major f16 attnD[bh][d 32][e 32].
__global__ void attn_softmax(const float* __restrict__ acc, const float* __restrict__ rescale,
                             _Float16* __restrict__ attnD)
{
    const int bh = blockIdx.x;
    const int h = bh % NHEAD;
    const int d = threadIdx.x;            // 0..31
    for (int e = 0; e < 32; ++e)
        attnD[(size_t)bh * 1024 + d * 32 + e] = (_Float16)0.f;
    if (d >= DHEAD) return;
    const float rs = rescale[h];
    float vals[DHEAD];
    float m = -1e30f;
    for (int e = 0; e < DHEAD; ++e) {
        const float v = acc[(size_t)bh * 1024 + d * 32 + e] * rs;
        vals[e] = v;
        m = fmaxf(m, v);
    }
    float s = 0.f;
    for (int e = 0; e < DHEAD; ++e) { vals[e] = expf(vals[e] - m); s += vals[e]; }
    const float inv = 1.f / s;
    for (int e = 0; e < DHEAD; ++e)
        attnD[(size_t)bh * 1024 + d * 32 + e] = (_Float16)(vals[e] * inv);
}

// o[r][h*31+d] = sum_e attn[d,e]*v[r][h,e]; all operands aligned vector loads.
// vbh: head-padded f16 [pix][64]; attnD zero-padded; obh: plain f16 [pix][64].
__global__ void attn_apply(const _Float16* __restrict__ vbh,
                           const _Float16* __restrict__ attnD,
                           _Float16* __restrict__ obh)
{
    const int lane = threadIdx.x & 31;
    const int wave = threadIdx.x >> 5;
    const int row0 = (blockIdx.x * 8 + wave) * 16;
    const int h = blockIdx.y >> 1, dt = blockIdx.y & 1;
    const int b = row0 / NPIX;
    const int bh = b * NHEAD + h;
    const int arow = row0 + (lane & 15);
    const int akb = (lane >> 4) * 16;
    const int bkb = (lane >> 4) * 16;
    const int dcol = dt * 16 + (lane & 15);

    const _Float16* ap = vbh + (size_t)arow * CP + h * 32;
    v16h a = combine(*(const v8h*)(ap + akb), *(const v8h*)(ap + akb + 8));
    const _Float16* bp = attnD + (size_t)bh * 1024 + dcol * 32;
    v16h bb = combine(*(const v8h*)(bp + bkb), *(const v8h*)(bp + bkb + 8));
    v8f c8 = {0.f,0.f,0.f,0.f,0.f,0.f,0.f,0.f};
    c8 = wmma16(a, bb, c8);

    const int mg = (lane >> 4) * 8;
    const int dd = dt * 16 + (lane & 15);
    if (dd >= DHEAD) return;
#pragma unroll
    for (int r = 0; r < 8; ++r)
        obh[(size_t)(row0 + mg + r) * CP + h * DHEAD + dd] = (_Float16)c8[r];
}

// ---------------------------------------------------------------------------
// Final 3x3 conv as 9-tap WMMA GEMM over pre-packed f16 input pk3[pix][128]
// (ch 0..61 = 1x1-conv output, 62..123 = z, pads x zero-padded weights).
// ---------------------------------------------------------------------------
__global__ void conv3x3_final(const _Float16* __restrict__ pk3,
                              const _Float16* __restrict__ W3t,  // [64][1152]
                              const float* __restrict__ bias, float* __restrict__ outN)
{
    const int lane = threadIdx.x & 31;
    const int wave = threadIdx.x >> 5;
    const int row0 = (blockIdx.x * 8 + wave) * 16;
    const int arow = row0 + (lane & 15);
    const int b = arow / NPIX;
    const int pp = arow % NPIX;
    const int ay = pp / WDIM, ax = pp % WDIM;
    const int akb = (lane >> 4) * 16;
    const int bkb = (lane >> 4) * 16;
    const int col = blockIdx.y * 16 + (lane & 15);

    const _Float16* wp = W3t + (size_t)col * KCONV;
    const v8h zero8 = {};
    v8f acc = {0.f,0.f,0.f,0.f,0.f,0.f,0.f,0.f};
    for (int tap = 0; tap < 9; ++tap) {
        const int ky = tap / 3, kx = tap % 3;
        const int yy = ay + ky - 1, xx = ax + kx - 1;
        const bool valid = (yy >= 0 && yy < HDIM && xx >= 0 && xx < WDIM);
        const _Float16* ap = pk3 + ((size_t)b * NPIX + yy * WDIM + xx) * C2P;
#pragma unroll
        for (int kk = 0; kk < C2P; kk += 32) {
            v8h alo = zero8, ahi = zero8;
            if (valid) {
                alo = *(const v8h*)(ap + kk + akb);
                ahi = *(const v8h*)(ap + kk + akb + 8);
            }
            v16h a = combine(alo, ahi);
            v16h bb = combine(*(const v8h*)(wp + tap * C2P + kk + bkb),
                              *(const v8h*)(wp + tap * C2P + kk + bkb + 8));
            acc = wmma16(a, bb, acc);
        }
    }
    const int mg = (lane >> 4) * 8;
    const int co = blockIdx.y * 16 + (lane & 15);
    if (co >= CDIM) return;
    const float bv = bias[co];
#pragma unroll
    for (int r = 0; r < 8; ++r) {
        const int rr = row0 + mg + r;
        const int b2 = rr / NPIX;
        const int p2 = rr % NPIX;
        outN[((size_t)b2 * CDIM + co) * NPIX + p2] = acc[r] + bv;
    }
}

// ---------------------------------------------------------------------------
// Elementwise / reduction kernels (bandwidth-bound paths)
// ---------------------------------------------------------------------------
__global__ void colnorm(const float* __restrict__ x, float* __restrict__ norms)
{
    __shared__ float red[256];
    const int bc = blockIdx.x;
    const int b = bc / CDIM, c = bc % CDIM;
    const float* base = x + (size_t)b * NPIX * CDIM + c;
    float s = 0.f;
    for (int p = threadIdx.x; p < NPIX; p += blockDim.x) {
        const float v = base[(size_t)p * CDIM];
        s += v * v;
    }
    red[threadIdx.x] = s;
    __syncthreads();
    for (int off = 128; off > 0; off >>= 1) {
        if ((int)threadIdx.x < off) red[threadIdx.x] += red[threadIdx.x + off];
        __syncthreads();
    }
    if (threadIdx.x == 0) norms[bc] = fmaxf(sqrtf(red[0]), 1e-12f);
}

// f32 [pix][62] -> head-padded f16 [pix][64] (slot h*32+d), divided by norm
__global__ void l2scale(const float* __restrict__ x, const float* __restrict__ norms,
                        _Float16* __restrict__ out)
{
    const size_t i = (size_t)blockIdx.x * blockDim.x + threadIdx.x;
    if (i >= (size_t)RTOT * CDIM) return;
    const int c = (int)(i % CDIM);
    const size_t r = i / CDIM;
    const int b = (int)(r / NPIX);
    out[r * CP + c + (c >= DHEAD ? 1 : 0)] = (_Float16)(x[i] / norms[b * CDIM + c]);
}

// Depthwise 3x3 SAME, f32 NHWC in/out; optional gelu / adds / f16 mirror.
__global__ void dwconv3x3_f(const float* __restrict__ in, const float* __restrict__ kw,
                            const float* __restrict__ add0, const float* __restrict__ add1,
                            float* __restrict__ out, _Float16* __restrict__ outH,
                            int gelu_out)
{
    const size_t i = (size_t)blockIdx.x * blockDim.x + threadIdx.x;
    if (i >= (size_t)RTOT * CDIM) return;
    const int c = (int)(i % CDIM);
    const size_t pix = i / CDIM;
    const int x = (int)(pix % WDIM);
    const int y = (int)((pix / WDIM) % HDIM);
    const int b = (int)(pix / NPIX);
    float s = 0.f;
#pragma unroll
    for (int ky = 0; ky < 3; ++ky)
#pragma unroll
        for (int kx = 0; kx < 3; ++kx) {
            const int yy = y + ky - 1, xx = x + kx - 1;
            if (yy >= 0 && yy < HDIM && xx >= 0 && xx < WDIM)
                s += in[((size_t)b * NPIX + yy * WDIM + xx) * CDIM + c] * kw[c * 9 + ky * 3 + kx];
        }
    if (gelu_out) s = gelu_exact(s);
    if (add0) s += add0[i];
    if (add1) s += add1[i];
    out[i] = s;
    if (outH) outH[pix * CP + c] = (_Float16)s;
}

// Depthwise 3x3 SAME on f16 hidden [pix][256] (c<248), fused GELU.
__global__ void dwconv3x3_h(const _Float16* __restrict__ in, const float* __restrict__ kw,
                            _Float16* __restrict__ out)
{
    const size_t i = (size_t)blockIdx.x * blockDim.x + threadIdx.x;
    if (i >= (size_t)RTOT * MDIM) return;
    const int c = (int)(i % MDIM);
    const size_t pix = i / MDIM;
    const int x = (int)(pix % WDIM);
    const int y = (int)((pix / WDIM) % HDIM);
    const int b = (int)(pix / NPIX);
    float s = 0.f;
#pragma unroll
    for (int ky = 0; ky < 3; ++ky)
#pragma unroll
        for (int kx = 0; kx < 3; ++kx) {
            const int yy = y + ky - 1, xx = x + kx - 1;
            if (yy >= 0 && yy < HDIM && xx >= 0 && xx < WDIM)
                s += (float)in[((size_t)b * NPIX + yy * WDIM + xx) * MP + c] *
                     kw[c * 9 + ky * 3 + kx];
        }
    out[pix * MP + c] = (_Float16)gelu_exact(s);
}

// LayerNorm over channels; writes f16 padded [pix][64].
__global__ void layernorm_k(const float* __restrict__ x, const float* __restrict__ g,
                            const float* __restrict__ bt, _Float16* __restrict__ outH)
{
    const size_t r = (size_t)blockIdx.x * blockDim.x + threadIdx.x;
    if (r >= (size_t)RTOT) return;
    const float* p = x + r * CDIM;
    float mu = 0.f;
    for (int c = 0; c < CDIM; ++c) mu += p[c];
    mu /= (float)CDIM;
    float var = 0.f;
    for (int c = 0; c < CDIM; ++c) { const float d = p[c] - mu; var += d * d; }
    var /= (float)CDIM;
    const float inv = rsqrtf(var + 1e-5f);
    for (int c = 0; c < CDIM; ++c)
        outH[r * CP + c] = (_Float16)((p[c] - mu) * inv * g[c] + bt[c]);
}

// NCHW f32 -> NHWC f32 (optional) + padded f16 mirror [pix][64]
__global__ void nchw_to_nhwc(const float* __restrict__ src, float* __restrict__ dstF,
                             _Float16* __restrict__ dstH)
{
    const size_t i = (size_t)blockIdx.x * blockDim.x + threadIdx.x;
    if (i >= (size_t)RTOT * CDIM) return;
    const int p = (int)(i % NPIX);
    const int c = (int)((i / NPIX) % CDIM);
    const int b = (int)(i / ((size_t)NPIX * CDIM));
    const float v = src[i];
    const size_t r = (size_t)b * NPIX + p;
    if (dstF) dstF[r * CDIM + c] = v;
    dstH[r * CP + c] = (_Float16)v;
}

// pack concat(a,b) (f32 [pix][62] each) -> f16 [pix][128]
__global__ void pack2h(const float* __restrict__ a, const float* __restrict__ b,
                       _Float16* __restrict__ dst)
{
    const size_t i = (size_t)blockIdx.x * blockDim.x + threadIdx.x;
    if (i >= (size_t)RTOT * 2 * CDIM) return;
    const int c = (int)(i % (2 * CDIM));
    const size_t r = i / (2 * CDIM);
    const float v = (c < CDIM) ? a[r * CDIM + c] : b[r * CDIM + (c - CDIM)];
    dst[r * C2P + c] = (_Float16)v;
}

// z (NCHW f32) -> pk3 channels 62..123
__global__ void z_to_pk3(const float* __restrict__ z, _Float16* __restrict__ pk3)
{
    const size_t i = (size_t)blockIdx.x * blockDim.x + threadIdx.x;
    if (i >= (size_t)RTOT * CDIM) return;
    const int p = (int)(i % NPIX);
    const int c = (int)((i / NPIX) % CDIM);
    const int b = (int)(i / ((size_t)NPIX * CDIM));
    pk3[((size_t)b * NPIX + p) * C2P + CDIM + c] = (_Float16)z[i];
}

__global__ void zero_f32(float* __restrict__ p, int n)
{
    const int i = blockIdx.x * blockDim.x + threadIdx.x;
    if (i < n) p[i] = 0.f;
}

// ------------------- weight conversion (transposed, zero-padded) -------------------
// src [K x Co] -> Wt [CoP x KP]
__global__ void cvt_w_t(const float* __restrict__ src, _Float16* __restrict__ dst,
                        int K, int Co, int KP, int CoP)
{
    const int i = blockIdx.x * blockDim.x + threadIdx.x;
    if (i >= CoP * KP) return;
    const int co = i / KP, k = i % KP;
    dst[i] = (k < K && co < Co) ? (_Float16)src[(size_t)k * Co + co] : (_Float16)0.f;
}
// src [Co x K] ('oi' conv1x1) -> Wt [CoP x KP]  (direct copy layout)
__global__ void cvt_oi_t(const float* __restrict__ src, _Float16* __restrict__ dst,
                         int K, int Co, int KP, int CoP)
{
    const int i = blockIdx.x * blockDim.x + threadIdx.x;
    if (i >= CoP * KP) return;
    const int co = i / KP, k = i % KP;
    dst[i] = (k < K && co < Co) ? (_Float16)src[(size_t)co * K + k] : (_Float16)0.f;
}
// src [Co x 124 x 3 x 3] -> W3t [64 x 1152], k = tap*128 + ci
__global__ void cvt_conv3_t(const float* __restrict__ src, _Float16* __restrict__ dst)
{
    const int i = blockIdx.x * blockDim.x + threadIdx.x;
    if (i >= CP * KCONV) return;
    const int co = i / KCONV, k = i % KCONV;
    const int tap = k / C2P, ci = k % C2P;
    dst[i] = (co < CDIM && ci < 2 * CDIM)
           ? (_Float16)src[((size_t)co * (2 * CDIM) + ci) * 9 + tap] : (_Float16)0.f;
}

// ===========================================================================
extern "C" void kernel_launch(void* const* d_in, const int* in_sizes, int n_in,
                              void* d_out, int out_size, void* d_ws, size_t ws_size,
                              hipStream_t stream)
{
    (void)in_sizes; (void)n_in; (void)out_size; (void)ws_size;

    const float* x       = (const float*)d_in[0];
    const float* y       = (const float*)d_in[1];
    const float* z       = (const float*)d_in[2];
    const float* wq_all  = (const float*)d_in[3];
    const float* wk_all  = (const float*)d_in[4];
    const float* wv_all  = (const float*)d_in[5];
    const float* rescale = (const float*)d_in[6];
    const float* pw_all  = (const float*)d_in[7];
    const float* pb_all  = (const float*)d_in[8];
    const float* dw1_all = (const float*)d_in[9];
    const float* dw2_all = (const float*)d_in[10];
    const float* ln_g    = (const float*)d_in[11];
    const float* ln_b    = (const float*)d_in[12];
    const float* ffw1    = (const float*)d_in[13];
    const float* ffdw    = (const float*)d_in[14];
    const float* ffw2    = (const float*)d_in[15];
    const float* fuc1_w  = (const float*)d_in[16];
    const float* fuc1_b  = (const float*)d_in[17];
    const float* fuc2_w  = (const float*)d_in[18];
    const float* fuc2_b  = (const float*)d_in[19];
    const float* outc_w  = (const float*)d_in[20];
    const float* outc_b  = (const float*)d_in[21];
    const float* final_w = (const float*)d_in[22];
    const float* final_b = (const float*)d_in[23];

    // ----------------- workspace bump allocator -----------------
    char* wp = (char*)d_ws;
    auto alloc = [&](size_t bytes) -> void* {
        void* r = (void*)wp;
        wp += (bytes + 255) & ~(size_t)255;
        return r;
    };
    const size_t SF = (size_t)RTOT * CDIM * 4;   // f32 activation
    const size_t SH = (size_t)RTOT * CP * 2;     // padded f16 mirror

    float* xn   = (float*)alloc(SF);
    float* yn   = (float*)alloc(SF);
    float* xa   = (float*)alloc(SF);
    float* yb   = (float*)alloc(SF);
    float* fa   = (float*)alloc(SF);
    float* fb   = (float*)alloc(SF);
    float* qb   = (float*)alloc(SF);
    float* kb   = (float*)alloc(SF);
    float* vb   = (float*)alloc(SF);
    float* t1   = (float*)alloc(SF);
    float* t2   = (float*)alloc(SF);
    float* tbuf = (float*)alloc(SF);
    float* fa2  = (float*)alloc(SF);
    float* fb2  = (float*)alloc(SF);

    _Float16* xnh = (_Float16*)alloc(SH);
    _Float16* ynh = (_Float16*)alloc(SH);
    _Float16* znh = (_Float16*)alloc(SH);
    _Float16* xah = (_Float16*)alloc(SH);
    _Float16* ybh = (_Float16*)alloc(SH);
    _Float16* fah = (_Float16*)alloc(SH);
    _Float16* fbh = (_Float16*)alloc(SH);
    _Float16* vbh = (_Float16*)alloc(SH);       // head-padded
    _Float16* qnh = (_Float16*)alloc(SH);       // head-padded
    _Float16* knh = (_Float16*)alloc(SH);       // head-padded
    _Float16* obh = (_Float16*)alloc(SH);
    _Float16* t1h = (_Float16*)alloc(SH);
    _Float16* hid1 = (_Float16*)alloc((size_t)RTOT * MP * 2);
    _Float16* hid2 = (_Float16*)alloc((size_t)RTOT * MP * 2);
    _Float16* packh = (_Float16*)alloc((size_t)RTOT * C2P * 2);
    _Float16* pk3   = (_Float16*)alloc((size_t)RTOT * C2P * 2);

    _Float16* wqt = (_Float16*)alloc((size_t)6 * CP * CP * 2);
    _Float16* wkt = (_Float16*)alloc((size_t)6 * CP * CP * 2);
    _Float16* wvt = (_Float16*)alloc((size_t)6 * CP * CP * 2);
    _Float16* pwt = (_Float16*)alloc((size_t)6 * CP * CP * 2);
    _Float16* w1t = (_Float16*)alloc((size_t)2 * MP * CP * 2);   // [256][64]
    _Float16* w2t = (_Float16*)alloc((size_t)2 * CP * MP * 2);   // [64][256]
    _Float16* fu1t = (_Float16*)alloc((size_t)CP * C2P * 2);
    _Float16* fu2t = (_Float16*)alloc((size_t)CP * C2P * 2);
    _Float16* outct = (_Float16*)alloc((size_t)CP * C2P * 2);
    _Float16* w3t  = (_Float16*)alloc((size_t)CP * KCONV * 2);
    _Float16* attnD = (_Float16*)alloc((size_t)BATCH * NHEAD * 1024 * 2);

    float* attn_acc = (float*)alloc((size_t)BATCH * NHEAD * 1024 * 4);
    float* normsQ   = (float*)alloc((size_t)BATCH * CDIM * 4);
    float* normsK   = (float*)alloc((size_t)BATCH * CDIM * 4);

    // ----------------- weight conversions -----------------
    const int wcc = CP * CP;
    for (int i = 0; i < 6; ++i) {
        cvt_w_t<<<(wcc + 255) / 256, 256, 0, stream>>>(wq_all + (size_t)i * CDIM * CDIM,
                                                       wqt + (size_t)i * wcc, CDIM, CDIM, CP, CP);
        cvt_w_t<<<(wcc + 255) / 256, 256, 0, stream>>>(wk_all + (size_t)i * CDIM * CDIM,
                                                       wkt + (size_t)i * wcc, CDIM, CDIM, CP, CP);
        cvt_w_t<<<(wcc + 255) / 256, 256, 0, stream>>>(wv_all + (size_t)i * CDIM * CDIM,
                                                       wvt + (size_t)i * wcc, CDIM, CDIM, CP, CP);
        cvt_w_t<<<(wcc + 255) / 256, 256, 0, stream>>>(pw_all + (size_t)i * CDIM * CDIM,
                                                       pwt + (size_t)i * wcc, CDIM, CDIM, CP, CP);
    }
    for (int j = 0; j < 2; ++j) {
        cvt_w_t<<<(MP * CP + 255) / 256, 256, 0, stream>>>(ffw1 + (size_t)j * CDIM * MDIM,
                                                           w1t + (size_t)j * MP * CP,
                                                           CDIM, MDIM, CP, MP);
        cvt_w_t<<<(CP * MP + 255) / 256, 256, 0, stream>>>(ffw2 + (size_t)j * MDIM * CDIM,
                                                           w2t + (size_t)j * CP * MP,
                                                           MDIM, CDIM, MP, CP);
    }
    cvt_oi_t<<<(CP * C2P + 255) / 256, 256, 0, stream>>>(fuc1_w, fu1t, 2 * CDIM, CDIM, C2P, CP);
    cvt_oi_t<<<(CP * C2P + 255) / 256, 256, 0, stream>>>(fuc2_w, fu2t, 2 * CDIM, CDIM, C2P, CP);
    cvt_oi_t<<<(CP * C2P + 255) / 256, 256, 0, stream>>>(outc_w, outct, 2 * CDIM, CDIM, C2P, CP);
    cvt_conv3_t<<<(CP * KCONV + 255) / 256, 256, 0, stream>>>(final_w, w3t);

    // ----------------- layout conversion -----------------
    const int nelem = RTOT * CDIM;
    const int eblk = (nelem + 255) / 256;
    nchw_to_nhwc<<<eblk, 256, 0, stream>>>(x, xn, xnh);
    nchw_to_nhwc<<<eblk, 256, 0, stream>>>(y, yn, ynh);
    nchw_to_nhwc<<<eblk, 256, 0, stream>>>(z, nullptr, znh);
    z_to_pk3<<<eblk, 256, 0, stream>>>(z, pk3);

    // ----------------- launch helpers -----------------
    auto gemm = [&](const _Float16* A, const _Float16* Wt, const float* bias,
                    const float* resid, float* outF, _Float16* outH,
                    int KP, int Co, int CoP, int strideH, int headpad, int dogelu) {
        dim3 g(RTOT / 128, CoP / 16);
        gemm_h<<<g, 256, 0, stream>>>(A, Wt, bias, resid, outF, outH,
                                      KP, Co, strideH, headpad, dogelu);
    };

    auto msa = [&](int i, const _Float16* qsh, const _Float16* ksh, const _Float16* vsh,
                   const float* resid, float* dst, _Float16* dsth) {
        const _Float16* wq = wqt + (size_t)i * wcc;
        const _Float16* wk = wkt + (size_t)i * wcc;
        const _Float16* wv = wvt + (size_t)i * wcc;
        const _Float16* pw = pwt + (size_t)i * wcc;
        gemm(qsh, wq, nullptr, nullptr, qb, nullptr, CP, CDIM, CP, 0, 0, 0);
        gemm(ksh, wk, nullptr, nullptr, kb, nullptr, CP, CDIM, CP, 0, 0, 0);
        gemm(vsh, wv, nullptr, nullptr, vb, vbh,     CP, CDIM, CP, CP, 1, 0);
        colnorm<<<BATCH * CDIM, 256, 0, stream>>>(qb, normsQ);
        colnorm<<<BATCH * CDIM, 256, 0, stream>>>(kb, normsK);
        l2scale<<<eblk, 256, 0, stream>>>(qb, normsQ, qnh);
        l2scale<<<eblk, 256, 0, stream>>>(kb, normsK, knh);
        zero_f32<<<(BATCH * NHEAD * 1024 + 255) / 256, 256, 0, stream>>>(
            attn_acc, BATCH * NHEAD * 1024);
        attn_partial<<<dim3(NPIX / 4096, BATCH * NHEAD), 128, 0, stream>>>(
            knh, qnh, attn_acc, 4096);
        attn_softmax<<<BATCH * NHEAD, 32, 0, stream>>>(attn_acc, rescale + i * NHEAD, attnD);
        attn_apply<<<dim3(RTOT / 128, NHEAD * 2), 256, 0, stream>>>(vbh, attnD, obh);
        gemm(obh, pw, pb_all + i * CDIM, nullptr, t1, nullptr, CP, CDIM, CP, 0, 0, 0);
        dwconv3x3_f<<<eblk, 256, 0, stream>>>(vb, dw1_all + (size_t)i * CDIM * 9,
                                              nullptr, nullptr, t2, nullptr, 1);
        dwconv3x3_f<<<eblk, 256, 0, stream>>>(t2, dw2_all + (size_t)i * CDIM * 9,
                                              t1, resid, dst, dsth, 0);
    };

    auto ffn = [&](int j, float* t) {
        layernorm_k<<<RTOT / 256, 256, 0, stream>>>(t, ln_g + j * CDIM, ln_b + j * CDIM, t1h);
        gemm(t1h, w1t + (size_t)j * MP * CP, nullptr, nullptr, nullptr, hid1,
             CP, MDIM, MP, MP, 0, 1);
        dwconv3x3_h<<<(RTOT * MDIM + 255) / 256, 256, 0, stream>>>(
            hid1, ffdw + (size_t)j * MDIM * 9, hid2);
        gemm(hid2, w2t + (size_t)j * CP * MP, nullptr, t, t, nullptr,
             MP, CDIM, CP, 0, 0, 0);
    };

    // ========================= pipeline =========================
    msa(0, xnh, xnh, xnh, xn, xa, xah);    // MSAB_1
    msa(1, ynh, ynh, ynh, yn, yb, ybh);    // MSAB_2
    msa(2, xnh, znh, znh, xn, fa, fah);    // LIGI_1
    msa(3, ynh, znh, znh, yn, fb, fbh);    // LIGI_2

    // GIGI_1
    msa(4, xah, fah, fah, xa, tbuf, nullptr);
    ffn(0, tbuf);
    pack2h<<<(2 * nelem + 255) / 256, 256, 0, stream>>>(tbuf, xa, packh);
    gemm(packh, fu1t, fuc1_b, nullptr, fa2, nullptr, C2P, CDIM, CP, 0, 0, 0);

    // GIGI_2
    msa(5, ybh, fbh, fbh, yb, tbuf, nullptr);
    ffn(1, tbuf);
    pack2h<<<(2 * nelem + 255) / 256, 256, 0, stream>>>(tbuf, yb, packh);
    gemm(packh, fu2t, fuc2_b, nullptr, fb2, nullptr, C2P, CDIM, CP, 0, 0, 0);

    // out_conv 1x1 (writes f16 directly into pk3 channels 0..61)
    pack2h<<<(2 * nelem + 255) / 256, 256, 0, stream>>>(fa2, fb2, packh);
    gemm(packh, outct, outc_b, nullptr, nullptr, pk3, C2P, CDIM, CP, C2P, 0, 0);

    // final 3x3 conv -> NCHW f32 output
    conv3x3_final<<<dim3(RTOT / 128, 4), 256, 0, stream>>>(pk3, w3t, final_b,
                                                           (float*)d_out);
}
